// BoundaryPredictor2_45363444580705
// MI455X (gfx1250) — matrix operand order repaired and checked by reading, not compiled, not run on hardware
//
#include <hip/hip_runtime.h>
#include <math.h>

#define B_ 8
#define S_ 4096
#define D_ 1024
#define SCALE_ 0.08838834764831843f   /* (1024/8)^-0.5 */
#define PRIOR_ 0.2f

typedef __attribute__((ext_vector_type(16))) __bf16 v16bf;
typedef __attribute__((ext_vector_type(8)))  float  v8f;
typedef __attribute__((ext_vector_type(4)))  int    v4i;
typedef __attribute__((ext_vector_type(8)))  int    v8i;
typedef __attribute__((ext_vector_type(4)))  unsigned int v4u;

union FragBF  { v16bf v; v4i d[2]; };
union Cvt8    { v4i v; unsigned short s[8]; };

#if defined(__has_builtin)
#  if __has_builtin(__builtin_amdgcn_tensor_load_to_lds) && \
      __has_builtin(__builtin_amdgcn_s_wait_tensorcnt)
#    define HAVE_TDM 1
#  else
#    define HAVE_TDM 0
#  endif
#else
#  define HAVE_TDM 0
#endif

__device__ __forceinline__ unsigned short f2bf(float f) {
  unsigned int u = __float_as_uint(f);
  u += 0x7FFFu + ((u >> 16) & 1u);      // round-to-nearest-even
  return (unsigned short)(u >> 16);
}

#if HAVE_TDM
// ---------------------------------------------------------------------------
// TDM: DMA a 2D bf16 tile (rows x tileK, row stride strideElems) from global
// into LDS at lds_addr.  D# built per CDNA5 ISA 8.3/8.4:
//   g0: [1:0]=count=1, [63:32]=lds_addr, [120:64]=global_addr, [127:126]=type2
//   g1: [17:16]=data_size(1=2B), [79:48]=tensor_dim0, [111:80]=tensor_dim1,
//       [127:112]=tile_dim0, [143:128]=tile_dim1, [207:160]=dim0_stride
// ---------------------------------------------------------------------------
__device__ __forceinline__ void tdm_load_2d(unsigned lds_addr,
                                            const unsigned short* gptr,
                                            int tileK, int rows,
                                            int strideElems)
{
  unsigned long long ga = (unsigned long long)(const void*)gptr;
  v4u g0;
  g0[0] = 1u;                                              // count=1
  g0[1] = lds_addr;                                        // lds byte addr
  g0[2] = (unsigned)ga;                                    // global[31:0]
  g0[3] = (unsigned)((ga >> 32) & 0x01FFFFFFull) | (2u << 30); // +type=2
  v8i g1;
  g1[0] = (1 << 16);                                       // data_size=2B
  g1[1] = (tileK & 0xFFFF) << 16;                          // tensor_dim0 lo
  g1[2] = ((tileK >> 16) & 0xFFFF) | ((rows & 0xFFFF) << 16); // d0 hi | d1 lo
  g1[3] = ((rows >> 16) & 0xFFFF) | ((tileK & 0xFFFF) << 16); // d1 hi | tile_dim0
  g1[4] = (rows & 0xFFFF);                                 // tile_dim1, tile_dim2=0
  g1[5] = strideElems;                                     // dim0_stride lo32
  g1[6] = 0;
  g1[7] = 0;
  v4i z4 = {0, 0, 0, 0};
#if defined(__clang_major__) && (__clang_major__ >= 23)
  v8i z8 = {0, 0, 0, 0, 0, 0, 0, 0};
  __builtin_amdgcn_tensor_load_to_lds(g0, g1, z4, z4, z8, 0);
#else
  __builtin_amdgcn_tensor_load_to_lds(g0, g1, z4, z4, 0);
#endif
}
#endif

// fallback / generic staging of a 1024x32 bf16 slice (row stride D_) into LDS
__device__ __forceinline__ void stage_slice(unsigned short* dst,
                                            const unsigned short* src,
                                            int tid)
{
  #pragma unroll
  for (int g = 0; g < 4; ++g) {
    v4i tmp[4];
    #pragma unroll
    for (int j = 0; j < 4; ++j) {
      int c = tid + (g * 4 + j) * 256;          // 0..4095 chunks of 8 elems
      int row = c >> 2, ch = c & 3;
      tmp[j] = *(const v4i*)(src + (size_t)row * D_ + ch * 8);
    }
    #pragma unroll
    for (int j = 0; j < 4; ++j) {
      int c = tid + (g * 4 + j) * 256;
      *(v4i*)(dst + c * 8) = tmp[j];            // LDS row-major [row][32]
    }
  }
}

// ---------------------------------------------------------------------------
// Kernel 1: h = (hidden + pos_emb) / max(||.||, 1e-12); write f32 + bf16 copies
// ---------------------------------------------------------------------------
__global__ void __launch_bounds__(256)
prep_kernel(const float* __restrict__ hidden, const float* __restrict__ pos,
            float* __restrict__ hf, unsigned short* __restrict__ hbf)
{
  int row = blockIdx.x;            // 0 .. B*S-1
  int s   = row & (S_ - 1);
  int tid = threadIdx.x;
  const float* hin = hidden + (size_t)row * D_ + tid * 4;
  const float* pin = pos    + (size_t)s   * D_ + tid * 4;
  float x0 = hin[0] + pin[0], x1 = hin[1] + pin[1];
  float x2 = hin[2] + pin[2], x3 = hin[3] + pin[3];
  float ss = x0*x0 + x1*x1 + x2*x2 + x3*x3;
  #pragma unroll
  for (int m = 1; m < 32; m <<= 1) ss += __shfl_xor(ss, m, 32);
  __shared__ float wsum[8];
  __shared__ float sinv;
  int lane = tid & 31, wv = tid >> 5;
  if (lane == 0) wsum[wv] = ss;
  __syncthreads();
  if (tid == 0) {
    float t = 0.f;
    #pragma unroll
    for (int i = 0; i < 8; ++i) t += wsum[i];
    sinv = 1.0f / fmaxf(sqrtf(t), 1e-12f);
  }
  __syncthreads();
  float inv = sinv;
  x0 *= inv; x1 *= inv; x2 *= inv; x3 *= inv;
  float* ho = hf + (size_t)row * D_ + tid * 4;
  unsigned short* bo = hbf + (size_t)row * D_ + tid * 4;
  ho[0] = x0; ho[1] = x1; ho[2] = x2; ho[3] = x3;
  bo[0] = f2bf(x0); bo[1] = f2bf(x1); bo[2] = f2bf(x2); bo[3] = f2bf(x3);
}

// ---------------------------------------------------------------------------
// Kernel 2: convert Wq / Wk to bf16
// ---------------------------------------------------------------------------
__global__ void __launch_bounds__(256)
cvt_kernel(const float* __restrict__ wq, const float* __restrict__ wk,
           unsigned short* __restrict__ wqb, unsigned short* __restrict__ wkb)
{
  int i = blockIdx.x * 256 + threadIdx.x;
  if (i < D_ * D_) wqb[i] = f2bf(wq[i]);
  else             wkb[i - D_ * D_] = f2bf(wk[i - D_ * D_]);
}

// ---------------------------------------------------------------------------
// Kernel 3: M[e,f] = sum_d Wq[d,e] * Wk[d,f] (bf16 out).  Contraction over the
// ROW index of both weights -> stage transposed tiles in LDS.  Staging is
// batched: one b128 global load per 8 elems into regs, then 8 ds_store_b16.
// ---------------------------------------------------------------------------
__global__ void __launch_bounds__(256)
mgemm_kernel(const unsigned short* __restrict__ wq,
             const unsigned short* __restrict__ wk,
             unsigned short* __restrict__ mbf)
{
  __shared__ unsigned short qT[16][32];    // [e_local][d_local]
  __shared__ unsigned short kT[512][32];   // [f_local][d_local]
  int tid  = threadIdx.x;
  int lane = tid & 31, wv = tid >> 5;
  int e0    = blockIdx.x * 16;
  int fbase = blockIdx.y * 512;
  int nl   = lane & 15;
  int half = lane >> 4;

  v8f zacc = {0.f,0.f,0.f,0.f,0.f,0.f,0.f,0.f};
  v8f acc[4];
  #pragma unroll
  for (int t = 0; t < 4; ++t) acc[t] = zacc;

  for (int k0 = 0; k0 < D_; k0 += 32) {
    if (tid < 64) {                                // qT: 64 chunks of 8
      int d = tid >> 1, e8 = (tid & 1) * 8;
      Cvt8 c; c.v = *(const v4i*)(wq + (size_t)(k0 + d) * D_ + e0 + e8);
      #pragma unroll
      for (int l = 0; l < 8; ++l) qT[e8 + l][d] = c.s[l];
    }
    {                                              // kT: 8 chunks per thread
      v4i tk[8];
      #pragma unroll
      for (int j = 0; j < 8; ++j) {
        int c = tid + j * 256;
        int d = c >> 6, f8 = (c & 63) * 8;
        tk[j] = *(const v4i*)(wk + (size_t)(k0 + d) * D_ + fbase + f8);
      }
      #pragma unroll
      for (int j = 0; j < 8; ++j) {
        int c = tid + j * 256;
        int d = c >> 6, f8 = (c & 63) * 8;
        Cvt8 u; u.v = tk[j];
        #pragma unroll
        for (int l = 0; l < 8; ++l) kT[f8 + l][d] = u.s[l];
      }
    }
    __syncthreads();

    FragBF fa;
    fa.d[0] = *(const v4i*)&qT[nl][16 * half];
    fa.d[1] = *(const v4i*)&qT[nl][16 * half + 8];
    #pragma unroll
    for (int t = 0; t < 4; ++t) {
      const unsigned short* bp = &kT[wv * 64 + t * 16 + nl][16 * half];
      FragBF fb;
      fb.d[0] = *(const v4i*)bp;
      fb.d[1] = *(const v4i*)(bp + 8);
      acc[t] = __builtin_amdgcn_wmma_f32_16x16x32_bf16(
          false, fa.v, false, fb.v, (short)0, acc[t], false, false);
    }
    __syncthreads();
  }
  #pragma unroll
  for (int t = 0; t < 4; ++t) {
    int f = fbase + wv * 64 + t * 16 + nl;
    #pragma unroll
    for (int r = 0; r < 8; ++r) {
      int e = e0 + 8 * half + r;
      mbf[(size_t)e * D_ + f] = f2bf(acc[t][r]);
    }
  }
}

// ---------------------------------------------------------------------------
// Kernel 4: score[b,s+1] = SCALE * h[b,s+1]^T (M h[b,s]) ; score[b,0]=0.
// Per WG: 32 s-rows x full 1024 e-cols, 8 waves (2 A-tiles x 8 B-tiles each).
// B operand (1024x32 bf16 M-slice per k-step) is DMA'd into LDS by the Tensor
// Data Mover, double-buffered (wave 0 issues, s_wait_tensorcnt + barrier),
// so all 8 waves share one staged copy instead of 8x L2 reads.
// A fragments come straight from global as b128s; dot with h[s+1] is fused
// into the epilogue (shuffle reduce + LDS across waves).
// ---------------------------------------------------------------------------
__global__ void __launch_bounds__(256)
adj_kernel(const unsigned short* __restrict__ hbf,
           const float* __restrict__ hf,
           const unsigned short* __restrict__ mbf,
           float* __restrict__ score)
{
  __shared__ __align__(16) unsigned short bstage[2][1024 * 32];
  __shared__ float red[8][32];

  int tid  = threadIdx.x;
  int lane = tid & 31, wv = tid >> 5;
  int b  = blockIdx.y;
  int s0 = blockIdx.x * 32;
  int colBase = wv * 128;
  int nl = lane & 15, half = lane >> 4;

  const unsigned short* rowA0 = hbf + ((size_t)b * S_ + s0 + nl) * D_;
  const unsigned short* rowA1 = rowA0 + (size_t)16 * D_;

  v8f zacc = {0.f,0.f,0.f,0.f,0.f,0.f,0.f,0.f};
  v8f acc0[8], acc1[8];
  #pragma unroll
  for (int t = 0; t < 8; ++t) { acc0[t] = zacc; acc1[t] = zacc; }

  // prologue: stage k-slice 0 into buffer 0
#if HAVE_TDM
  if (wv == 0)
    tdm_load_2d((unsigned)(size_t)(const void*)&bstage[0][0], mbf, 32, 1024, D_);
#else
  stage_slice(&bstage[0][0], mbf, tid);
#endif

  for (int it = 0; it < 32; ++it) {
    int k0  = it * 32;
    int cur = it & 1;
#if HAVE_TDM
    if (wv == 0) __builtin_amdgcn_s_wait_tensorcnt(0);
#endif
    __syncthreads();                       // bstage[cur] ready, [1-cur] free
    if (it + 1 < 32) {
#if HAVE_TDM
      if (wv == 0)
        tdm_load_2d((unsigned)(size_t)(const void*)&bstage[1 - cur][0],
                    mbf + (it + 1) * 32, 32, 1024, D_);
#else
      stage_slice(&bstage[1 - cur][0], mbf + (it + 1) * 32, tid);
#endif
    }

    FragBF a0, a1;
    a0.d[0] = *(const v4i*)(rowA0 + k0 + 16 * half);
    a0.d[1] = *(const v4i*)(rowA0 + k0 + 16 * half + 8);
    a1.d[0] = *(const v4i*)(rowA1 + k0 + 16 * half);
    a1.d[1] = *(const v4i*)(rowA1 + k0 + 16 * half + 8);
    if (k0 + 32 < D_) {                    // global_prefetch_b8 next A tile
      __builtin_prefetch(rowA0 + k0 + 32, 0, 3);
      __builtin_prefetch(rowA1 + k0 + 32, 0, 3);
    }
    const unsigned short* bb = &bstage[cur][0];
    #pragma unroll
    for (int t = 0; t < 8; ++t) {
      const unsigned short* bp = bb + (colBase + t * 16 + nl) * 32 + 16 * half;
      FragBF fb;
      fb.d[0] = *(const v4i*)bp;
      fb.d[1] = *(const v4i*)(bp + 8);
      acc0[t] = __builtin_amdgcn_wmma_f32_16x16x32_bf16(
          false, a0.v, false, fb.v, (short)0, acc0[t], false, false);
      acc1[t] = __builtin_amdgcn_wmma_f32_16x16x32_bf16(
          false, a1.v, false, fb.v, (short)0, acc1[t], false, false);
    }
  }

  // epilogue: p[m] += D[m,n] * h[b, s+1, e]
  float p0[8], p1[8];
  #pragma unroll
  for (int r = 0; r < 8; ++r) { p0[r] = 0.f; p1[r] = 0.f; }
  const float* hfb = hf + (size_t)b * S_ * D_;
  #pragma unroll
  for (int t = 0; t < 8; ++t) {
    int e = colBase + t * 16 + nl;
    #pragma unroll
    for (int r = 0; r < 8; ++r) {
      int row0 = s0 + 8 * half + r;
      int row1 = row0 + 16;
      if (row0 + 1 < S_) p0[r] += acc0[t][r] * hfb[(size_t)(row0 + 1) * D_ + e];
      if (row1 + 1 < S_) p1[r] += acc1[t][r] * hfb[(size_t)(row1 + 1) * D_ + e];
    }
  }
  #pragma unroll
  for (int r = 0; r < 8; ++r) {
    #pragma unroll
    for (int msk = 1; msk < 16; msk <<= 1) {
      p0[r] += __shfl_xor(p0[r], msk, 32);
      p1[r] += __shfl_xor(p1[r], msk, 32);
    }
  }
  if ((lane & 15) == 0) {
    #pragma unroll
    for (int r = 0; r < 8; ++r) {
      red[wv][8 * half + r]      = p0[r];
      red[wv][16 + 8 * half + r] = p1[r];
    }
  }
  __syncthreads();
  if (tid < 32) {
    float t = 0.f;
    #pragma unroll
    for (int w = 0; w < 8; ++w) t += red[w][tid];
    int s = s0 + tid;
    if (s + 1 < S_) score[(size_t)b * S_ + s + 1] = SCALE_ * t;
  }
  if (tid == 0 && s0 == 0) score[(size_t)b * S_] = 0.f;
}

// ---------------------------------------------------------------------------
// Kernel 5: probs -> gumbel-sigmoid -> hard -> cumsum -> segment ids + k[b]
// ---------------------------------------------------------------------------
__global__ void __launch_bounds__(256)
scan_kernel(const float* __restrict__ score, const float* __restrict__ noise,
            const float* __restrict__ sig_temp, const float* __restrict__ sig_thr,
            int* __restrict__ seg, float* __restrict__ kcnt)
{
  int b = blockIdx.x;
  int tid = threadIdx.x;
  float temp = sig_temp[0];
  float thr  = sig_thr[0];
  const float eps = 1.1920929e-7f;
  float vals[16];
  float run = 0.f;
  #pragma unroll
  for (int j = 0; j < 16; ++j) {
    int s = tid * 16 + j;
    float sc = score[(size_t)b * S_ + s];
    float pr = 1.f / (1.f + expf(-(sc - thr) / temp));
    pr = fminf(fmaxf(pr, 0.f), 1.f);
    if (s == 0) pr = 1.f;
    float p  = fminf(fmaxf(pr, eps), 1.f - eps);
    float lo = logf(p) - log1pf(-p);
    float u  = fminf(fmaxf(noise[(size_t)b * S_ + s], eps), 1.f - eps);
    float lg = logf(u) - log1pf(-u);
    float soft = 1.f / (1.f + expf(-(lo + lg)));       // TEMP = 1
    float hard = soft > 0.5f ? 1.f : 0.f;
    run += hard;
    vals[j] = run;
  }
  __shared__ float tsum[256];
  __shared__ float texcl[256];
  tsum[tid] = run;
  __syncthreads();
  if (tid == 0) {
    float a = 0.f;
    for (int i = 0; i < 256; ++i) { texcl[i] = a; a += tsum[i]; }
    kcnt[b] = a;
  }
  __syncthreads();
  float base = texcl[tid];
  #pragma unroll
  for (int j = 0; j < 16; ++j) {
    int c = (int)(base + vals[j] + 0.5f) - 1;
    c = c < 0 ? 0 : (c > S_ - 1 ? S_ - 1 : c);
    seg[(size_t)b * S_ + tid * 16 + j] = c;
  }
}

// ---------------------------------------------------------------------------
// Kernel 6: segment mean-pool into pooled[S,B,D] (output pre-zeroed)
// ---------------------------------------------------------------------------
__global__ void __launch_bounds__(256)
pool_kernel(const float* __restrict__ hf, const int* __restrict__ seg,
            float* __restrict__ out)
{
  int b = blockIdx.y;
  int s = blockIdx.x;
  const int* sb = seg + (size_t)b * S_;
  int id = sb[s];
  if (s > 0 && sb[s - 1] == id) return;      // not a segment start
  int e = s + 1;
  while (e < S_ && sb[e] == id) ++e;
  float inv = 1.f / (float)(e - s);
  int d0 = threadIdx.x * 4;
  float a0 = 0.f, a1 = 0.f, a2 = 0.f, a3 = 0.f;
  for (int r = s; r < e; ++r) {
    const float* p = hf + ((size_t)b * S_ + r) * D_ + d0;
    a0 += p[0]; a1 += p[1]; a2 += p[2]; a3 += p[3];
  }
  float* o = out + ((size_t)id * B_ + b) * D_ + d0;
  o[0] = a0 * inv; o[1] = a1 * inv; o[2] = a2 * inv; o[3] = a3 * inv;
}

// ---------------------------------------------------------------------------
// Kernel 7: binomial prior loss
// ---------------------------------------------------------------------------
__global__ void loss_kernel(const float* __restrict__ kcnt, float* __restrict__ out)
{
  const float n = (float)S_;
  float acc = 0.f;
  for (int b = 0; b < B_; ++b) {
    float k = kcnt[b];
    float lp = lgammaf(n + 1.f) - lgammaf(k + 1.f) - lgammaf(n - k + 1.f)
             + k * logf(PRIOR_) + (n - k) * log1pf(-PRIOR_);
    acc += lp;
  }
  out[(size_t)S_ * B_ * D_] = -(acc / (float)B_) / n;
}

// ---------------------------------------------------------------------------
extern "C" void kernel_launch(void* const* d_in, const int* in_sizes, int n_in,
                              void* d_out, int out_size, void* d_ws, size_t ws_size,
                              hipStream_t stream)
{
  const float* hidden   = (const float*)d_in[0];
  const float* pos_emb  = (const float*)d_in[1];
  const float* Wq       = (const float*)d_in[2];
  const float* Wk       = (const float*)d_in[3];
  const float* sig_temp = (const float*)d_in[4];
  const float* sig_thr  = (const float*)d_in[5];
  const float* noise_u  = (const float*)d_in[6];

  char* ws = (char*)d_ws;
  float*          hf   = (float*)(ws);                         // 134,217,728 B
  unsigned short* hbf  = (unsigned short*)(ws + 134217728ull); //  67,108,864 B
  unsigned short* wqb  = (unsigned short*)(ws + 201326592ull); //   2,097,152 B
  unsigned short* wkb  = (unsigned short*)(ws + 203423744ull); //   2,097,152 B
  unsigned short* mbf  = (unsigned short*)(ws + 205520896ull); //   2,097,152 B
  float*          score= (float*)(ws + 207618048ull);          //     131,072 B
  int*            seg  = (int*)  (ws + 207749120ull);          //     131,072 B
  float*          kcnt = (float*)(ws + 207880192ull);          //          32 B

  float* out = (float*)d_out;

  (void)hipMemsetAsync(d_out, 0, (size_t)out_size * sizeof(float), stream);
  prep_kernel <<<B_ * S_, 256, 0, stream>>>(hidden, pos_emb, hf, hbf);
  cvt_kernel  <<<(2 * D_ * D_) / 256, 256, 0, stream>>>(Wq, Wk, wqb, wkb);
  mgemm_kernel<<<dim3(64, 2), 256, 0, stream>>>(wqb, wkb, mbf);
  adj_kernel  <<<dim3(S_ / 32, B_), 256, 0, stream>>>(hbf, hf, mbf, score);
  scan_kernel <<<B_, 256, 0, stream>>>(score, noise_u, sig_temp, sig_thr, seg, kcnt);
  pool_kernel <<<dim3(S_, B_), 256, 0, stream>>>(hf, seg, out);
  loss_kernel <<<1, 1, 0, stream>>>(kcnt, out);
}
